// PointTransformerExtractor_86268713107539
// MI455X (gfx1250) — matrix operand order repaired and checked by reading, not compiled, not run on hardware
//
#include <hip/hip_runtime.h>
#include <hip/hip_bf16.h>
#include <math.h>

// ---------------------------------------------------------------------------
// PointTransformer extractor for MI455X (gfx1250, wave32, WMMA).
// Dense GEMMs and pt_conv edge-MLPs -> v_wmma_f32_16x16x32_f16.
// Irregular parts (kNN, FPS, softmax, gathers) -> scalar VALU, L2-resident.
// Wave indices go through readfirstlane so tile loops are SALU-controlled and
// EXEC is structurally all-ones at every WMMA (ISA 7.12 requirement).
// ---------------------------------------------------------------------------

#define KNN 16
#define BIGF 1e9f

typedef __attribute__((ext_vector_type(16))) _Float16 v16h;
typedef __attribute__((ext_vector_type(8)))  float    v8f;

union F16x16 { v16h v; _Float16 h[16]; };
union F32x8  { v8f  v; float    f[8];  };

// Build the A fragment (16x32 f16, ISA layout) for rows [base .. base+15] of a
// row-major f32 matrix with leading dim ldA, K-slice starting at k0.
__device__ __forceinline__ v16h frag_a(const float* __restrict__ A, int ldA,
                                       int rowbase, int k0, int lane) {
  int grp = lane >> 4, l16 = lane & 15;
  const float* rp = A + (size_t)(rowbase + l16) * ldA;
  F16x16 af;
#pragma unroll
  for (int e = 0; e < 16; ++e) {
    int ka = k0 + ((e < 8) ? e : (e + 8)) + (grp << 3);
    af.h[e] = (_Float16)rp[ka];
  }
  return af.v;
}

// Build the B fragment (32x16 f16, ISA layout) for cols [n0 .. n0+15] of a
// row-major f32 matrix with leading dim ldW, K-slice starting at k0.
__device__ __forceinline__ v16h frag_b(const float* __restrict__ W, int ldW,
                                       int n0, int k0, int lane) {
  int grp = lane >> 4, l16 = lane & 15;
  F16x16 bf;
#pragma unroll
  for (int e = 0; e < 16; ++e) {
    int kb = k0 + ((e < 8) ? e : (e + 8)) + (grp << 3);
    bf.h[e] = (_Float16)W[(size_t)kb * ldW + n0 + l16];
  }
  return bf.v;
}

// 16xK @ Kx16 -> 16x16 f32 accumulator (K % 32 == 0).
__device__ __forceinline__ F32x8 wmma_mm16(const float* __restrict__ A, int ldA,
                                           const float* __restrict__ W, int ldW,
                                           int n0, int K, int lane) {
  F32x8 acc;
#pragma unroll
  for (int g = 0; g < 8; ++g) acc.f[g] = 0.0f;
  for (int k0 = 0; k0 < K; k0 += 32) {
    v16h a = frag_a(A, ldA, 0, k0, lane);
    v16h b = frag_b(W, ldW, n0, k0, lane);
    acc.v = __builtin_amdgcn_wmma_f32_16x16x32_f16(false, a, false, b, (short)0,
                                                   acc.v, false, false);
  }
  return acc;
}

// ---------------------------------------------------------------------------
// Generic WMMA GEMM: out[M,N] = epilogue(A[M,K] @ W[K,N])
// MODE 0: none ; 1: +bias, relu ; 2: +bias, *gamma/sqrt(1+eps)+beta, relu
// M % 16 == 0, N % 32 == 0, K % 32 == 0 (holds at every call site).
// One wave computes a 16x32 strip (A fragment reused across 2 WMMAs).
// ---------------------------------------------------------------------------
template <int MODE>
__global__ void k_gemm_wmma(const float* __restrict__ A, const float* __restrict__ W,
                            const float* __restrict__ bias,
                            const float* __restrict__ gamma, const float* __restrict__ beta,
                            float* __restrict__ out,
                            int M, int N, int K, int strips) {
  // readfirstlane -> SGPR wave id: early-exit is a scalar branch, EXEC untouched.
  int wave = (blockIdx.x << 2) + __builtin_amdgcn_readfirstlane(threadIdx.x >> 5);
  if (wave >= strips) return;
  int stripsN = N >> 5;
  int tm = wave / stripsN;
  int n0 = (wave - tm * stripsN) << 5;

  int lane = threadIdx.x & 31;
  int grp  = lane >> 4;
  int l16  = lane & 15;

  F32x8 acc0, acc1;
#pragma unroll
  for (int g = 0; g < 8; ++g) { acc0.f[g] = 0.0f; acc1.f[g] = 0.0f; }

  for (int k0 = 0; k0 < K; k0 += 32) {
    v16h a  = frag_a(A, K, tm << 4, k0, lane);
    v16h b0 = frag_b(W, N, n0,      k0, lane);
    v16h b1 = frag_b(W, N, n0 + 16, k0, lane);
    acc0.v = __builtin_amdgcn_wmma_f32_16x16x32_f16(false, a, false, b0, (short)0,
                                                    acc0.v, false, false);
    acc1.v = __builtin_amdgcn_wmma_f32_16x16x32_f16(false, a, false, b1, (short)0,
                                                    acc1.v, false, false);
  }

  const float rs = 0.999995000037f;           // 1/sqrt(1 + 1e-5)
  int nA = n0 + l16, nB = n0 + 16 + l16;
  float bi0 = 0.f, bi1 = 0.f, ga0 = 1.f, ga1 = 1.f, be0 = 0.f, be1 = 0.f;
  if (MODE >= 1) { bi0 = bias[nA]; bi1 = bias[nB]; }
  if (MODE == 2) { ga0 = gamma[nA] * rs; ga1 = gamma[nB] * rs; be0 = beta[nA]; be1 = beta[nB]; }

#pragma unroll
  for (int g = 0; g < 8; ++g) {
    int m = (tm << 4) + g + (grp << 3);
    float v0 = acc0.f[g], v1 = acc1.f[g];
    if (MODE >= 1) { v0 += bi0; v1 += bi1; }
    if (MODE == 2) { v0 = v0 * ga0 + be0; v1 = v1 * ga1 + be1; }
    if (MODE >= 1) { v0 = fmaxf(v0, 0.f); v1 = fmaxf(v1, 0.f); }
    out[(size_t)m * N + nA] = v0;
    out[(size_t)m * N + nB] = v1;
  }
}

// ---------------------------------------------------------------------------
// Input MLP: y = relu(gamma/sqrt(1+eps) * (data@W[3,32] + b) + beta)
// ---------------------------------------------------------------------------
__global__ void k_input_mlp(const float* __restrict__ data, const float* __restrict__ W,
                            const float* __restrict__ b, const float* __restrict__ gamma,
                            const float* __restrict__ beta, float* __restrict__ out, int MP) {
  int t = blockIdx.x * blockDim.x + threadIdx.x;
  if (t >= MP * 32) return;
  int p = t >> 5, c = t & 31;
  const float* d = data + p * 3;
  float y = d[0] * W[c] + d[1] * W[32 + c] + d[2] * W[64 + c] + b[c];
  y = y * (gamma[c] * 0.999995000037f) + beta[c];
  out[t] = fmaxf(y, 0.0f);
}

// ---------------------------------------------------------------------------
// Register-resident predicated top-16 insertion (smallest distances first).
// ---------------------------------------------------------------------------
__device__ __forceinline__ void topk_insert(float d, int j, float bd[KNN], int bi[KNN]) {
  if (d >= bd[KNN - 1]) return;
  int pos = 0;
#pragma unroll
  for (int t = 0; t < KNN; ++t) pos += (bd[t] <= d) ? 1 : 0;
#pragma unroll
  for (int t = KNN - 1; t > 0; --t)
    if (t > pos) { bd[t] = bd[t - 1]; bi[t] = bi[t - 1]; }
#pragma unroll
  for (int t = 0; t < KNN; ++t)
    if (t == pos) { bd[t] = d; bi[t] = j; }
}

// kNN of pos against itself (self excluded). One thread per center.
__global__ void k_knn_self(const float* __restrict__ pos, int* __restrict__ nb, int B, int N) {
  int t = blockIdx.x * blockDim.x + threadIdx.x;
  if (t >= B * N) return;
  int b = t / N, i = t - b * N;
  const float* pb = pos + (size_t)b * N * 3;
  float cx = pb[i * 3], cy = pb[i * 3 + 1], cz = pb[i * 3 + 2];
  float bd[KNN]; int bi[KNN];
#pragma unroll
  for (int q = 0; q < KNN; ++q) { bd[q] = BIGF; bi[q] = 0; }
  for (int j = 0; j < N; ++j) {
    if (j == i) continue;
    float dx = pb[j * 3] - cx, dy = pb[j * 3 + 1] - cy, dz = pb[j * 3 + 2] - cz;
    topk_insert(dx * dx + dy * dy + dz * dz, j, bd, bi);
  }
#pragma unroll
  for (int q = 0; q < KNN; ++q) nb[t * KNN + q] = bi[q];
}

// kNN of sub_pos centers against pos. One thread per center.
__global__ void k_knn_cross(const float* __restrict__ sub_pos, const float* __restrict__ pos,
                            int* __restrict__ nb, int B, int Mq, int N) {
  int t = blockIdx.x * blockDim.x + threadIdx.x;
  if (t >= B * Mq) return;
  int b = t / Mq;
  const float* pb = pos + (size_t)b * N * 3;
  float cx = sub_pos[t * 3], cy = sub_pos[t * 3 + 1], cz = sub_pos[t * 3 + 2];
  float bd[KNN]; int bi[KNN];
#pragma unroll
  for (int q = 0; q < KNN; ++q) { bd[q] = BIGF; bi[q] = 0; }
  for (int j = 0; j < N; ++j) {
    float dx = pb[j * 3] - cx, dy = pb[j * 3 + 1] - cy, dz = pb[j * 3 + 2] - cz;
    topk_insert(dx * dx + dy * dy + dz * dz, j, bd, bi);
  }
#pragma unroll
  for (int q = 0; q < KNN; ++q) nb[t * KNN + q] = bi[q];
}

// ---------------------------------------------------------------------------
// Farthest point sampling: one block per batch, start at index 0 (deterministic).
// ---------------------------------------------------------------------------
__global__ void k_fps(const float* __restrict__ pos, int N, int m, int* __restrict__ idx) {
  __shared__ float mind[4096];
  __shared__ float rmax[256];
  __shared__ int   rarg[256];
  __shared__ int   s_cur;
  int b = blockIdx.x, tid = threadIdx.x;
  const float* pb = pos + (size_t)b * N * 3;
  for (int i = tid; i < N; i += 256) mind[i] = BIGF;
  if (tid == 0) s_cur = 0;
  __syncthreads();
  for (int it = 0; it < m; ++it) {
    int cur = s_cur;
    if (tid == 0) idx[b * m + it] = cur;
    float cx = pb[cur * 3], cy = pb[cur * 3 + 1], cz = pb[cur * 3 + 2];
    float bestv = -1.0f; int besti = 0;
    for (int i = tid; i < N; i += 256) {
      float dx = pb[i * 3] - cx, dy = pb[i * 3 + 1] - cy, dz = pb[i * 3 + 2] - cz;
      float d = dx * dx + dy * dy + dz * dz;
      float mv = fminf(mind[i], d);
      mind[i] = mv;
      if (mv > bestv) { bestv = mv; besti = i; }
    }
    rmax[tid] = bestv; rarg[tid] = besti;
    __syncthreads();
    for (int s = 128; s > 0; s >>= 1) {
      if (tid < s) {
        bool take = (rmax[tid + s] > rmax[tid]) ||
                    (rmax[tid + s] == rmax[tid] && rarg[tid + s] < rarg[tid]);
        if (take) { rmax[tid] = rmax[tid + s]; rarg[tid] = rarg[tid + s]; }
      }
      __syncthreads();
    }
    if (tid == 0) s_cur = rarg[0];
    __syncthreads();
  }
}

__global__ void k_gather_pos(const float* __restrict__ pos, const int* __restrict__ idx,
                             float* __restrict__ out, int B, int N, int m) {
  int t = blockIdx.x * blockDim.x + threadIdx.x;
  if (t >= B * m * 3) return;
  int d = t % 3, bm = t / 3, b = bm / m;
  out[t] = pos[((size_t)b * N + idx[bm]) * 3 + d];
}

// x_out[b,mm,c] = max_k xbn[b, nb[b,mm,k], c]
__global__ void k_gather_max(const float* __restrict__ xbn, const int* __restrict__ nb,
                             float* __restrict__ out, int BM, int C, int N, int m) {
  int t = blockIdx.x * blockDim.x + threadIdx.x;
  if (t >= BM * C) return;
  int c = t % C, bm = t / C, b = bm / m;
  const int* nbp = nb + bm * KNN;
  const float* xb = xbn + (size_t)b * N * C;
  float best = -BIGF;
#pragma unroll
  for (int q = 0; q < KNN; ++q) best = fmaxf(best, xb[(size_t)nbp[q] * C + c]);
  out[t] = best;
}

// ---------------------------------------------------------------------------
// Fused PointTransformerConv: one block (256 thr = 8 waves) per center point.
// Edge MLP inner GEMMs run on WMMA with LDS-staged A operands:
//   delta = relu(h[16,64]   @ pW2[64,C] + pb2)
//   h2    = relu(t[16,C]    @ aW1[C,64] + ab1)
//   alpha = relu(h2[16,64]  @ aW2[64,C] + ab2)
// Waves own 16-column output tiles; wave id is an SGPR (readfirstlane) so the
// tile loops are scalar-branch controlled and EXEC stays all-ones at WMMA.
// Then per-channel softmax over k=16 neighbors + weighted aggregation.
// ---------------------------------------------------------------------------
__global__ void k_ptconv(const float* __restrict__ v, const float* __restrict__ asrc,
                         const float* __restrict__ adst, const float* __restrict__ pos,
                         const int* __restrict__ nb,
                         const float* __restrict__ pW1, const float* __restrict__ pb1,
                         const float* __restrict__ pW2, const float* __restrict__ pb2,
                         const float* __restrict__ aW1, const float* __restrict__ ab1,
                         const float* __restrict__ aW2, const float* __restrict__ ab2,
                         float* __restrict__ out, int N, int C) {
  __shared__ float s_delta[KNN * 256];
  __shared__ float s_alpha[KNN * 256];   // also holds attn input "t"
  __shared__ float s_h[KNN * 64];        // hidden (64) for both edge MLPs
  __shared__ float s_adst[256];
  __shared__ float s_pm[KNN * 3];
  __shared__ int   s_jn[KNN];

  int blk = blockIdx.x;
  int b = blk / N, i = blk - b * N;
  int tid  = threadIdx.x;
  int wave = __builtin_amdgcn_readfirstlane(tid >> 5);   // SGPR wave id
  int lane = tid & 31;
  int grp  = lane >> 4, l16 = lane & 15;
  size_t rowC = ((size_t)b * N + i) * C;

  if (tid < KNN) s_jn[tid] = nb[((size_t)b * N + i) * KNN + tid];
  __syncthreads();
  if (tid < KNN * 3) {
    int j = tid / 3, d = tid - j * 3;
    s_pm[tid] = pos[((size_t)b * N + i) * 3 + d] - pos[((size_t)b * N + s_jn[j]) * 3 + d];
  }
  for (int c = tid; c < C; c += 256) s_adst[c] = adst[rowC + c];
  __syncthreads();

  // h = relu(pm @ pW1[3,64] + pb1)   (K=3: scalar)
  for (int t = tid; t < KNN * 64; t += 256) {
    int j = t >> 6, o = t & 63;
    float s = pb1[o] + s_pm[j * 3] * pW1[o] + s_pm[j * 3 + 1] * pW1[64 + o] +
              s_pm[j * 3 + 2] * pW1[128 + o];
    s_h[t] = fmaxf(s, 0.0f);
  }
  __syncthreads();

  // delta = relu(h @ pW2[64,C] + pb2)   -- WMMA, K=64
  for (int n0 = wave << 4; n0 < C; n0 += 128) {
    F32x8 r = wmma_mm16(s_h, 64, pW2, C, n0, 64, lane);
    int c = n0 + l16;
    float bi = pb2[c];
#pragma unroll
    for (int g = 0; g < 8; ++g) {
      int j = g + (grp << 3);
      s_delta[j * 256 + c] = fmaxf(r.f[g] + bi, 0.0f);
    }
  }
  __syncthreads();

  // t = adst_i - asrc_j + delta   (stored in s_alpha, row stride 256)
  for (int t = tid; t < KNN * C; t += 256) {
    int j = t / C, c = t - j * C;
    s_alpha[j * 256 + c] =
        s_adst[c] - asrc[((size_t)b * N + s_jn[j]) * C + c] + s_delta[j * 256 + c];
  }
  __syncthreads();

  // h2 = relu(t @ aW1[C,64] + ab1)   -- WMMA, K=C
  for (int n0 = wave << 4; n0 < 64; n0 += 128) {
    F32x8 r = wmma_mm16(s_alpha, 256, aW1, 64, n0, C, lane);
    int o = n0 + l16;
    float bi = ab1[o];
#pragma unroll
    for (int g = 0; g < 8; ++g) {
      int j = g + (grp << 3);
      s_h[j * 64 + o] = fmaxf(r.f[g] + bi, 0.0f);
    }
  }
  __syncthreads();

  // alpha = relu(h2 @ aW2[64,C] + ab2)   -- WMMA, K=64
  for (int n0 = wave << 4; n0 < C; n0 += 128) {
    F32x8 r = wmma_mm16(s_h, 64, aW2, C, n0, 64, lane);
    int c = n0 + l16;
    float bi = ab2[c];
#pragma unroll
    for (int g = 0; g < 8; ++g) {
      int j = g + (grp << 3);
      s_alpha[j * 256 + c] = fmaxf(r.f[g] + bi, 0.0f);
    }
  }
  __syncthreads();

  // per-channel softmax over 16 neighbors + aggregation
  for (int c = tid; c < C; c += 256) {
    float mx = -BIGF;
#pragma unroll
    for (int j = 0; j < KNN; ++j) mx = fmaxf(mx, s_alpha[j * 256 + c]);
    float ssum = 0.0f;
#pragma unroll
    for (int j = 0; j < KNN; ++j) ssum += expf(s_alpha[j * 256 + c] - mx);
    float acc = 0.0f;
#pragma unroll
    for (int j = 0; j < KNN; ++j) {
      float w = expf(s_alpha[j * 256 + c] - mx) / ssum;
      acc += w * (v[((size_t)b * N + s_jn[j]) * C + c] + s_delta[j * 256 + c]);
    }
    out[rowC + c] = acc;
  }
}

// mean over points: out[b,c] = mean_n x[b,n,c]
__global__ void k_meanpool(const float* __restrict__ x, float* __restrict__ out, int B, int N, int C) {
  int t = blockIdx.x * blockDim.x + threadIdx.x;
  if (t >= B * C) return;
  int b = t / C, c = t - b * C;
  float s = 0.0f;
  for (int n = 0; n < N; ++n) s += x[((size_t)b * N + n) * C + c];
  out[t] = s / (float)N;
}

// ---------------------------------------------------------------------------
// Host-side orchestration
// ---------------------------------------------------------------------------
struct Bump {
  char* base; size_t off;
  float* f(size_t n) { float* p = (float*)(base + off); off = (off + n * 4 + 255) & ~(size_t)255; return p; }
  int*   i(size_t n) { int*   p = (int*)  (base + off); off = (off + n * 4 + 255) & ~(size_t)255; return p; }
};

struct TB {
  const float *aW1, *ab1, *aW2, *ab2;   // attn_nn l1/l2
  const float *linW, *ldstW;            // lin.W, lin_dst.W
  const float *linW_in, *linb_in;       // lin_in
  const float *linW_out, *linb_out;     // lin_out
  const float *lsrcW;                   // lin_src.W
  const float *pW1, *pb1, *pW2, *pb2;   // pos_nn l1/l2
};

static TB get_tb(void* const* din, int b0) {
  TB t;
  t.aW1 = (const float*)din[b0 + 0];  t.ab1 = (const float*)din[b0 + 1];
  t.aW2 = (const float*)din[b0 + 2];  t.ab2 = (const float*)din[b0 + 3];
  t.linW = (const float*)din[b0 + 4]; t.ldstW = (const float*)din[b0 + 5];
  t.linW_in = (const float*)din[b0 + 6];  t.linb_in = (const float*)din[b0 + 7];
  t.linW_out = (const float*)din[b0 + 8]; t.linb_out = (const float*)din[b0 + 9];
  t.lsrcW = (const float*)din[b0 + 10];
  t.pW1 = (const float*)din[b0 + 11]; t.pb1 = (const float*)din[b0 + 12];
  t.pW2 = (const float*)din[b0 + 13]; t.pb2 = (const float*)din[b0 + 14];
  return t;
}

static void run_gemm(hipStream_t s, const float* A, const float* W, const float* bias,
                     const float* gamma, const float* beta, float* out,
                     int M, int N, int K, int mode) {
  int strips = (M / 16) * (N / 32);
  int blocks = (strips + 3) / 4;
  if (mode == 0)
    k_gemm_wmma<0><<<blocks, 128, 0, s>>>(A, W, bias, gamma, beta, out, M, N, K, strips);
  else if (mode == 1)
    k_gemm_wmma<1><<<blocks, 128, 0, s>>>(A, W, bias, gamma, beta, out, M, N, K, strips);
  else
    k_gemm_wmma<2><<<blocks, 128, 0, s>>>(A, W, bias, gamma, beta, out, M, N, K, strips);
}

static void run_tblock(hipStream_t s, Bump& bp, const TB& p, const float* x_in,
                       const float* pos, float* x_out, int B, int N, int C) {
  int M = B * N;
  float* xin = bp.f((size_t)M * C);
  run_gemm(s, x_in, p.linW_in, p.linb_in, nullptr, nullptr, xin, M, C, C, 1);

  int* nb = bp.i((size_t)M * KNN);
  k_knn_self<<<(M + 127) / 128, 128, 0, s>>>(pos, nb, B, N);

  float* v    = bp.f((size_t)M * C);
  float* asrc = bp.f((size_t)M * C);
  float* adst = bp.f((size_t)M * C);
  run_gemm(s, xin, p.linW,  nullptr, nullptr, nullptr, v,    M, C, C, 0);
  run_gemm(s, xin, p.lsrcW, nullptr, nullptr, nullptr, asrc, M, C, C, 0);
  run_gemm(s, xin, p.ldstW, nullptr, nullptr, nullptr, adst, M, C, C, 0);

  float* y = bp.f((size_t)M * C);
  k_ptconv<<<M, 256, 0, s>>>(v, asrc, adst, pos, nb,
                             p.pW1, p.pb1, p.pW2, p.pb2,
                             p.aW1, p.ab1, p.aW2, p.ab2, y, N, C);

  run_gemm(s, y, p.linW_out, p.linb_out, nullptr, nullptr, x_out, M, C, C, 1);
}

extern "C" void kernel_launch(void* const* d_in, const int* in_sizes, int n_in,
                              void* d_out, int out_size, void* d_ws, size_t ws_size,
                              hipStream_t stream) {
  (void)in_sizes; (void)n_in; (void)out_size; (void)ws_size;
  const int B = 4;
  const float* data = (const float*)d_in[0];    // [4,4096,3]

  // mlp_input (bn_mlp): beta, gamma, lin.W, lin.b at indices 1..4
  const float* in_beta  = (const float*)d_in[1];
  const float* in_gamma = (const float*)d_in[2];
  const float* in_W     = (const float*)d_in[3];
  const float* in_b     = (const float*)d_in[4];

  TB t_input = get_tb(d_in, 5);
  TB tb0 = get_tb(d_in, 20);
  TB tb1 = get_tb(d_in, 35);
  TB tb2 = get_tb(d_in, 50);
  // td bn_mlps: beta, gamma, lin.W, lin.b
  const float* tdB[3]  = { (const float*)d_in[65], (const float*)d_in[69], (const float*)d_in[73] };
  const float* tdG[3]  = { (const float*)d_in[66], (const float*)d_in[70], (const float*)d_in[74] };
  const float* tdW[3]  = { (const float*)d_in[67], (const float*)d_in[71], (const float*)d_in[75] };
  const float* tdb[3]  = { (const float*)d_in[68], (const float*)d_in[72], (const float*)d_in[76] };

  const int Ns[4] = {4096, 1024, 256, 64};
  const int Cs[4] = {32, 64, 128, 256};
  const TB* tbs[3] = { &tb0, &tb1, &tb2 };

  Bump bp{(char*)d_ws, 0};

  // --- input MLP: [B*4096,3] -> [B*4096,32]
  int M0 = B * Ns[0];
  float* x = bp.f((size_t)M0 * Cs[0]);
  k_input_mlp<<<(M0 * 32 + 255) / 256, 256, 0, stream>>>(data, in_W, in_b, in_gamma, in_beta, x, M0);

  // --- t_input transformer block at level 0
  float* x0 = bp.f((size_t)M0 * Cs[0]);
  run_tblock(stream, bp, t_input, x, data, x0, B, Ns[0], Cs[0]);

  const float* pos = data;
  const float* xcur = x0;

  for (int lvl = 0; lvl < 3; ++lvl) {
    int Np = Ns[lvl], Cn = Cs[lvl + 1], Cp = Cs[lvl];
    int m = Ns[lvl + 1];

    // transition down
    int* fidx = bp.i((size_t)B * m);
    k_fps<<<B, 256, 0, stream>>>(pos, Np, m, fidx);

    float* pos_new = bp.f((size_t)B * m * 3);
    k_gather_pos<<<(B * m * 3 + 255) / 256, 256, 0, stream>>>(pos, fidx, pos_new, B, Np, m);

    int* nbc = bp.i((size_t)B * m * KNN);
    k_knn_cross<<<(B * m + 127) / 128, 128, 0, stream>>>(pos_new, pos, nbc, B, m, Np);

    float* xbn = bp.f((size_t)B * Np * Cn);
    run_gemm(stream, xcur, tdW[lvl], tdb[lvl], tdG[lvl], tdB[lvl], xbn, B * Np, Cn, Cp, 2);

    float* xdown = bp.f((size_t)B * m * Cn);
    k_gather_max<<<(B * m * Cn + 255) / 256, 256, 0, stream>>>(xbn, nbc, xdown, B * m, Cn, Np, m);

    // transformer block at new level
    float* xnew = bp.f((size_t)B * m * Cn);
    run_tblock(stream, bp, *tbs[lvl], xdown, pos_new, xnew, B, m, Cn);

    pos = pos_new;
    xcur = xnew;
  }

  // --- global mean pool over the final 64 points -> [4,256]
  k_meanpool<<<(B * Cs[3] + 255) / 256, 256, 0, stream>>>(xcur, (float*)d_out, B, Ns[3], Cs[3]);
}
